// ParallelDecoder_82463372083235
// MI455X (gfx1250) — matrix operand order repaired and checked by reading, not compiled, not run on hardware
//
#include <hip/hip_runtime.h>
#include <hip/hip_bf16.h>
#include <utility>

typedef __attribute__((ext_vector_type(16))) __bf16 v16bf;
typedef __attribute__((ext_vector_type(8)))  __bf16 v8bf;
typedef __attribute__((ext_vector_type(8)))  float  v8f;

#define H 2048
#define G 8192            // 4*H gate rows
#define O 1024
#define SEQ 256

// ---------------------------------------------------------------------------
// WMMA matvec inner loop: acc += W[rowbase:rowbase+16, k_begin:k_end] * h
// W row-major [*, 2048] bf16; h replicated into all 16 B-columns so every
// column of D holds the same y; we later extract column 0.
// A-frag (16-bit 16x32): lane m<16 -> row m, K {k0..k0+15};
// lane m+16 -> row m, K {k0+16..k0+31}. Two 16B loads per lane.
// B-frag: lanes 0-15 need h[k0..k0+15], lanes 16-31 need h[k0+16..k0+31]
// (independent of column since all columns are identical): one 32B load.
// ---------------------------------------------------------------------------
__device__ inline v8f wmma_matvec(const __bf16* __restrict__ W,
                                  const __bf16* __restrict__ h,
                                  int rowbase, int k_begin, int k_end, v8f acc) {
  const int lane = threadIdx.x & 31;
  const int m    = lane & 15;
  const int hi   = lane >> 4;
  const __bf16* wrow = W + (size_t)(rowbase + m) * H + hi * 16;
  for (int k0 = k_begin; k0 < k_end; k0 += 32) {
    v8bf alo = *(const v8bf*)(wrow + k0);
    v8bf ahi = *(const v8bf*)(wrow + k0 + 8);
    v16bf a  = __builtin_shufflevector(alo, ahi,
                 0,1,2,3,4,5,6,7,8,9,10,11,12,13,14,15);
    v16bf b  = *(const v16bf*)(h + k0 + hi * 16);
    acc = __builtin_amdgcn_wmma_f32_16x16x32_bf16(
        /*neg_a=*/false, a, /*neg_b=*/false, b,
        /*c_mod=*/(short)0, acc, /*reuse_a=*/false, /*reuse_b=*/false);
  }
  return acc;
}

// ---------------------------------------------------------------------------
// One LSTM layer step for 16 gate indices per block.
// grid = 128 blocks (H/16), block = 256 threads (8 waves).
// wave -> (tile = gate group i/f/g/o, half = K half). nmats==2 adds the
// W_ih * h_below matvec (layer 1). base = precomputed bias(+x) term [8192].
// ---------------------------------------------------------------------------
__global__ __launch_bounds__(256) void lstm_step_kernel(
    const __bf16* __restrict__ Wa, const __bf16* __restrict__ ha,
    const __bf16* __restrict__ Wb, const __bf16* __restrict__ hb,
    const float* __restrict__ base, float* __restrict__ c,
    __bf16* __restrict__ h_out, int nmats)
{
  __shared__ float red[2][4 * 16];
  const int wave = threadIdx.x >> 5;
  const int lane = threadIdx.x & 31;
  const int tile = wave & 3;      // gate group: 0=i 1=f 2=g 3=o
  const int half = wave >> 2;     // K half
  const int rowbase = tile * H + blockIdx.x * 16;

  v8f acc = {};
  acc = wmma_matvec(Wa, ha, rowbase, half * 1024, half * 1024 + 1024, acc);
  if (nmats > 1)
    acc = wmma_matvec(Wb, hb, rowbase, half * 1024, half * 1024 + 1024, acc);

  // D column 0 lives in lane 0 (M=0..7 in vgpr 0..7) and lane 16 (M=8..15).
  if (lane == 0) {
#pragma unroll
    for (int r = 0; r < 8; ++r) red[half][tile * 16 + r] = acc[r];
  }
  if (lane == 16) {
#pragma unroll
    for (int r = 0; r < 8; ++r) red[half][tile * 16 + 8 + r] = acc[r];
  }
  __syncthreads();

  if (threadIdx.x < 16) {
    const int tid = threadIdx.x;
    const int j   = blockIdx.x * 16 + tid;
    float gi = base[0 * H + j] + red[0][0 * 16 + tid] + red[1][0 * 16 + tid];
    float gf = base[1 * H + j] + red[0][1 * 16 + tid] + red[1][1 * 16 + tid];
    float gg = base[2 * H + j] + red[0][2 * 16 + tid] + red[1][2 * 16 + tid];
    float go = base[3 * H + j] + red[0][3 * 16 + tid] + red[1][3 * 16 + tid];
    float si = 1.f / (1.f + __expf(-gi));
    float sf = 1.f / (1.f + __expf(-gf));
    float so = 1.f / (1.f + __expf(-go));
    float cn = sf * c[j] + si * tanhf(gg);
    c[j] = cn;                         // private to this thread across blocks
    h_out[j] = (__bf16)(so * tanhf(cn));
  }
}

// ---------------------------------------------------------------------------
// out = leaky_relu(W_out @ h + b_out), written to row (255 - step).
// grid = 64 blocks (O/16), 8 waves each owning a K chunk of 256.
// ---------------------------------------------------------------------------
__global__ __launch_bounds__(256) void out_proj_kernel(
    const __bf16* __restrict__ W, const __bf16* __restrict__ h,
    const float* __restrict__ bias, float* __restrict__ out)
{
  __shared__ float red[8][16];
  const int wave = threadIdx.x >> 5;
  const int lane = threadIdx.x & 31;

  v8f acc = {};
  acc = wmma_matvec(W, h, blockIdx.x * 16, wave * 256, wave * 256 + 256, acc);

  if (lane == 0) {
#pragma unroll
    for (int r = 0; r < 8; ++r) red[wave][r] = acc[r];
  }
  if (lane == 16) {
#pragma unroll
    for (int r = 0; r < 8; ++r) red[wave][8 + r] = acc[r];
  }
  __syncthreads();

  if (threadIdx.x < 16) {
    const int j = blockIdx.x * 16 + threadIdx.x;
    float s = bias[j];
#pragma unroll
    for (int w = 0; w < 8; ++w) s += red[w][threadIdx.x];
    out[j] = s > 0.f ? s : 0.01f * s;   // leaky_relu slope 0.01
  }
}

// ---------------------------------------------------------------------------
// One-time: fp32 -> bf16 weight conversion (L2-resident working set).
// ---------------------------------------------------------------------------
__global__ void convert_kernel(__bf16* __restrict__ dst,
                               const float* __restrict__ src, int n) {
  int i = blockIdx.x * blockDim.x + threadIdx.x;
  int stride = gridDim.x * blockDim.x;
  for (; i < n; i += stride) dst[i] = (__bf16)src[i];
}

// One-time: base0[row] = b_ih0[row] + b_hh0[row] + dot(w_ih0[row,:], x).
// 1024 blocks * 8 waves = 8192 rows; wave-level shuffle reduction.
__global__ __launch_bounds__(256) void base0_kernel(
    float* __restrict__ base0, const float* __restrict__ w_ih0,
    const float* __restrict__ x, const float* __restrict__ b_ih,
    const float* __restrict__ b_hh)
{
  const int wave = threadIdx.x >> 5;
  const int lane = threadIdx.x & 31;
  const int row  = blockIdx.x * 8 + wave;
  const float* wr = w_ih0 + (size_t)row * H;
  float s = 0.f;
  for (int k = lane; k < H; k += 32) s += wr[k] * x[k];
#pragma unroll
  for (int off = 16; off; off >>= 1) s += __shfl_down(s, off, 32);
  if (lane == 0) base0[row] = s + b_ih[row] + b_hh[row];
}

// One-time: base1 = b_ih1 + b_hh1; copy initial h (as bf16) and c (fp32).
__global__ void init_state_kernel(
    float* __restrict__ base1, __bf16* __restrict__ h0buf,
    __bf16* __restrict__ h1buf, float* __restrict__ c0buf,
    float* __restrict__ c1buf, const float* __restrict__ h0,
    const float* __restrict__ c0, const float* __restrict__ b_ih,
    const float* __restrict__ b_hh)
{
  int i = blockIdx.x * blockDim.x + threadIdx.x;
  if (i < G) base1[i] = b_ih[G + i] + b_hh[G + i];
  if (i < H) {
    h0buf[i] = (__bf16)h0[i];
    h1buf[i] = (__bf16)h0[H + i];
    c0buf[i] = c0[i];
    c1buf[i] = c0[H + i];
  }
}

extern "C" void kernel_launch(void* const* d_in, const int* in_sizes, int n_in,
                              void* d_out, int out_size, void* d_ws, size_t ws_size,
                              hipStream_t stream) {
  const float* x     = (const float*)d_in[0];   // [1,1,H]
  const float* h0    = (const float*)d_in[1];   // [2,1,H]
  const float* c0    = (const float*)d_in[2];   // [2,1,H]
  const float* w_ih  = (const float*)d_in[3];   // [2,4H,H]
  const float* w_hh  = (const float*)d_in[4];   // [2,4H,H]
  const float* b_ih  = (const float*)d_in[5];   // [2,4H]
  const float* b_hh  = (const float*)d_in[6];   // [2,4H]
  const float* w_out = (const float*)d_in[7];   // [O,H]
  const float* b_out = (const float*)d_in[8];   // [O]
  float* out = (float*)d_out;                   // [SEQ,1,O]

  // ---- workspace carving (256B aligned) ----
  char* p = (char*)d_ws;
  auto carve = [&](size_t bytes) {
    char* r = p;
    p += (bytes + 255) & ~(size_t)255;
    return r;
  };
  const size_t WG = (size_t)G * H;  // 16.78M elements per recurrent matrix
  __bf16* wb_hh0 = (__bf16*)carve(WG * sizeof(__bf16));
  __bf16* wb_ih1 = (__bf16*)carve(WG * sizeof(__bf16));
  __bf16* wb_hh1 = (__bf16*)carve(WG * sizeof(__bf16));
  __bf16* wb_out = (__bf16*)carve((size_t)O * H * sizeof(__bf16));
  float*  base0  = (float*)carve(G * sizeof(float));
  float*  base1  = (float*)carve(G * sizeof(float));
  float*  c0buf  = (float*)carve(H * sizeof(float));
  float*  c1buf  = (float*)carve(H * sizeof(float));
  __bf16* h0bufA = (__bf16*)carve(H * sizeof(__bf16));
  __bf16* h0bufB = (__bf16*)carve(H * sizeof(__bf16));
  __bf16* h1bufA = (__bf16*)carve(H * sizeof(__bf16));
  __bf16* h1bufB = (__bf16*)carve(H * sizeof(__bf16));

  // ---- one-time setup (runs every call -> deterministic replays) ----
  convert_kernel<<<2048, 256, 0, stream>>>(wb_hh0, w_hh, (int)WG);
  convert_kernel<<<2048, 256, 0, stream>>>(wb_ih1, w_ih + WG, (int)WG);
  convert_kernel<<<2048, 256, 0, stream>>>(wb_hh1, w_hh + WG, (int)WG);
  convert_kernel<<<2048, 256, 0, stream>>>(wb_out, w_out, O * H);
  base0_kernel<<<G / 8, 256, 0, stream>>>(base0, w_ih, x, b_ih, b_hh);
  init_state_kernel<<<(G + 255) / 256, 256, 0, stream>>>(
      base1, h0bufA, h1bufA, c0buf, c1buf, h0, c0, b_ih, b_hh);

  // ---- 256 sequential steps ----
  __bf16* h0cur = h0bufA; __bf16* h0nxt = h0bufB;
  __bf16* h1cur = h1bufA; __bf16* h1nxt = h1bufB;
  for (int t = 0; t < SEQ; ++t) {
    // layer 0: gates = base0 + W_hh0 @ h0cur
    lstm_step_kernel<<<H / 16, 256, 0, stream>>>(
        wb_hh0, h0cur, nullptr, nullptr, base0, c0buf, h0nxt, 1);
    // layer 1: gates = base1 + W_ih1 @ h0nxt + W_hh1 @ h1cur
    lstm_step_kernel<<<H / 16, 256, 0, stream>>>(
        wb_ih1, h0nxt, wb_hh1, h1cur, base1, c1buf, h1nxt, 2);
    // projection, reversed output order
    out_proj_kernel<<<O / 16, 256, 0, stream>>>(
        wb_out, h1nxt, b_out, out + (size_t)(SEQ - 1 - t) * O);
    std::swap(h0cur, h0nxt);
    std::swap(h1cur, h1nxt);
  }
}
